// MMmKModule_2035814498575
// MI455X (gfx1250) — compile-verified, matrix operand order checked by reading, not run on hardware
//
#include <hip/hip_runtime.h>
#include <hip/hip_bf16.h>
#include <math.h>

typedef __attribute__((ext_vector_type(16))) _Float16 v16h;
typedef __attribute__((ext_vector_type(8)))  float    v8f;

#define K_STATES   4096
#define N_SERVERS  64
#define VMIN       1e-20f
#define VMAX       1.0f
#define LN64       4.1588830833596718565f   // ln(64)

// log of unnormalized stationary weight:
// l_n = n*ln(lambda/mu) - [ lgamma(min(n,64)+1) + max(0, n-64)*ln(64) ]
__device__ __forceinline__ float log_unnorm(int n, float r) {
    int m = (n < N_SERVERS) ? n : N_SERVERS;
    float S = lgammaf((float)(m + 1));
    if (n > N_SERVERS) S += (float)(n - N_SERVERS) * LN64;
    return (float)n * r - S;
}

__global__ __launch_bounds__(256)
void mmk_steady_kernel(const float* __restrict__ lambd,
                       const float* __restrict__ mu,
                       const int*   __restrict__ ind,
                       float*       __restrict__ out,
                       int nOut)
{
    __shared__ float s_partial[256];
    __shared__ float s_lnZ;

    const int   t = threadIdx.x;
    const float r = logf(lambd[0]) - logf(mu[0]);

    // ---- Phase A: per-thread partial sums of exp(l_n), 16 states each ----
    float p = 0.0f;
    #pragma unroll
    for (int i = 0; i < 16; ++i) {
        int n = t * 16 + i;
        p += expf(log_unnorm(n, r));       // bounded by ~e^2; no overflow
    }
    s_partial[t] = p;
    __syncthreads();

    // ---- Phase B: wave 0 reduces 256 partials; WMMA does the 64-way add ----
    if (t < 32) {                          // full wave => EXEC all ones (WMMA req.)
        // 8 partials per lane -> one fp32 value per lane (32 values)
        float v = 0.0f;
        #pragma unroll
        for (int j = 0; j < 8; ++j) v += s_partial[t * 8 + j];

        // hi/lo f16 split so the f16 WMMA reduction keeps ~fp32 accuracy:
        // v = hi + lo exactly to within 2^-22 relative.
        _Float16 hi = (_Float16)v;
        _Float16 lo = (_Float16)(v - (float)hi);

        v16h a, b;
        #pragma unroll
        for (int i = 0; i < 16; ++i) { a[i] = (_Float16)1.0f; b[i] = (_Float16)0.0f; }
        b[0] = hi;                          // 64 nonzero B elements total across wave
        b[1] = lo;

        v8f c = {};
        // D[m,n] = sum_k A[m,k]*B[k,n] = column-sum(B) (A is all ones),
        // identical for every row m. d[0] holds colsum[lane & 15] in each lane.
        v8f d = __builtin_amdgcn_wmma_f32_16x16x32_f16(
            /*neg_a=*/false, a, /*neg_b=*/false, b,
            /*c_mod=*/(short)0, c, /*reuse_a=*/false, /*reuse_b=*/false);

        // Fold the 16 column sums -> total Z (xor tree stays inside groups of 16).
        float total = d[0];
        #pragma unroll
        for (int off = 1; off < 16; off <<= 1)
            total += __shfl_xor(total, off, 32);

        if (t == 0) s_lnZ = logf(total);
    }
    __syncthreads();

    // ---- Phase C: evaluate only the requested states ----
    if (t < nOut) {
        int   n  = ind[t];
        float pi = expf(log_unnorm(n, r) - s_lnZ);   // underflow -> 0 -> clipped
        pi = fminf(fmaxf(pi, VMIN), VMAX);
        out[t] = pi;
    }
}

extern "C" void kernel_launch(void* const* d_in, const int* in_sizes, int n_in,
                              void* d_out, int out_size, void* d_ws, size_t ws_size,
                              hipStream_t stream) {
    const float* lambd = (const float*)d_in[0];   // [1] f32
    const float* mu    = (const float*)d_in[1];   // [1] f32
    const int*   ind   = (const int*)d_in[2];     // [256] i32
    float*       out   = (float*)d_out;           // [256] f32
    (void)n_in; (void)d_ws; (void)ws_size;

    mmk_steady_kernel<<<1, 256, 0, stream>>>(lambd, mu, ind, out, out_size);
}
